// Fractal2DNonDiff_65395172049480
// MI455X (gfx1250) — matrix operand order repaired and checked by reading, not compiled, not run on hardware
//
#include <hip/hip_runtime.h>
#include <hip/hip_bf16.h>

typedef __attribute__((ext_vector_type(16))) _Float16 v16h;
typedef __attribute__((ext_vector_type(8)))  _Float16 v8h;
typedef __attribute__((ext_vector_type(8)))  float    v8f;

#define NVD 40          // NUM_VALUES * NUM_DUPES
#define PPT 4           // points per thread (ILP to hide ds_load latency)
#define TPB 256

#define PAD_R 48        // 40 rows padded to 3x16
#define PAD_C 64        // 40 cols padded to 2x32

// ---------------------------------------------------------------------------
// Kernel 1: recover child indices from one-hot matrices via WMMA.
// argmax(one_hot_row) == one_hot_row . iota  (exact: 0/1 and k<64 are exact
// in f16; f32 accumulate). One wave, v_wmma_f32_16x16x32_f16.
// The matrix is staged as a zero-padded 48x64 f16 image in LDS so the A
// fragment is two aligned ds_load_b128 per tile (no predicated loads).
// ---------------------------------------------------------------------------
__global__ __launch_bounds__(32)
void child_idx_wmma(const float* __restrict__ Lm, const float* __restrict__ Rm,
                    int* __restrict__ children /* [2][NVD] */) {
    __shared__ alignas(16) _Float16 As[PAD_R * PAD_C];   // 6 KB

    const int lane      = threadIdx.x;      // 0..31
    const int half      = lane >> 4;        // 0: lanes 0-15, 1: lanes 16-31
    const int m_in_tile = lane & 15;

    for (int mat = 0; mat < 2; ++mat) {
        const float* M = (mat == 0) ? Lm : Rm;
        int* outp = children + mat * NVD;

        // zero-fill padded image (as dwords), then drop in the 40x40 values
        for (int i = lane; i < PAD_R * PAD_C / 2; i += 32)
            ((unsigned*)As)[i] = 0u;
        __syncthreads();
        for (int i = lane; i < NVD * NVD; i += 32) {
            int r = i / NVD, cc = i % NVD;
            As[r * PAD_C + cc] = (_Float16)M[i];
        }
        __syncthreads();

        for (int t = 0; t < 3; ++t) {               // row tiles: rows t*16..t*16+15
            v8f c;
            #pragma unroll
            for (int r = 0; r < 8; ++r) c[r] = 0.0f;

            #pragma unroll
            for (int kc = 0; kc < 2; ++kc) {        // K chunks: cols kc*32..kc*32+31
                // A 16x32 f16 layout: lane L (M = L&15), halves:
                //   j=0..7  -> K = kc*32 + half*8 + j
                //   j=8..15 -> K = kc*32 + 16 + half*8 + (j-8)
                const int row = t * 16 + m_in_tile;
                const _Float16* pa = &As[row * PAD_C + kc * 32 + half * 8];
                v8h a_lo = *(const v8h*)pa;         // 16B aligned
                v8h a_hi = *(const v8h*)(pa + 16);  // 16B aligned
                v16h a;
                #pragma unroll
                for (int j = 0; j < 8; ++j) { a[j] = a_lo[j]; a[8 + j] = a_hi[j]; }

                // B 32x16 f16 layout: lane L (N = L&15), j<8 -> K = half*8 + j, j>=8 -> K = 16 + half*8 + (j-8).
                // B[k][n] = global column index k (broadcast across N).
                v16h b;
                #pragma unroll
                for (int j = 0; j < 16; ++j)
                    b[j] = (_Float16)(float)(kc * 32 + half * 8 + j + (j >= 8 ? 8 : 0));

                c = __builtin_amdgcn_wmma_f32_16x16x32_f16(
                        false, a, false, b, (short)0, c, false, false);
            }
            // D layout: VGPR r, lanes 0-15 -> M=r,N=lane ; lanes 16-31 -> M=8+r,N=lane-16.
            // Take column N==0 (lanes 0 and 16).
            if (m_in_tile == 0) {
                #pragma unroll
                for (int r = 0; r < 8; ++r) {
                    int m = t * 16 + half * 8 + r;
                    if (m < NVD) outp[m] = (int)(c[r] + 0.5f);
                }
            }
        }
        __syncthreads();   // protect LDS image before next matrix overwrites it
    }
}

// ---------------------------------------------------------------------------
// Kernel 2: the traversal. Node packed into 8B in LDS: one ds_load_b64/step.
//   word0 = f32 split_point
//   word1 = (dir!=0)<<16 | left<<8 | right
// ---------------------------------------------------------------------------
__global__ __launch_bounds__(TPB)
void fractal_traverse(const float* __restrict__ X, const float* __restrict__ Y,
                      const float* __restrict__ sp, const float* __restrict__ sd,
                      const float* __restrict__ vals,
                      const int* __restrict__ children,
                      const int* __restrict__ max_depth_p,
                      float* __restrict__ out, int n) {
    __shared__ uint2 nodes[NVD];
    __shared__ float svals[NVD];

    const int tid = threadIdx.x;
    if (tid < NVD) {
        uint2 nd;
        nd.x = __float_as_uint(sp[tid]);
        unsigned dir = (sd[tid] != 0.0f) ? 1u : 0u;
        unsigned l   = (unsigned)children[tid];
        unsigned r   = (unsigned)children[NVD + tid];
        nd.y = (dir << 16) | (l << 8) | r;
        nodes[tid] = nd;
        svals[tid] = vals[tid];
    }
    __syncthreads();

    const int md   = *max_depth_p;
    const int base = blockIdx.x * (TPB * PPT) + tid;

    float x[PPT], y[PPT], minx[PPT], maxx[PPT], miny[PPT], maxy[PPT];
    int   idx[PPT];
    #pragma unroll
    for (int u = 0; u < PPT; ++u) {
        int p = base + u * TPB;
        bool ok = p < n;
        x[u] = ok ? X[p] : 0.0f;
        y[u] = ok ? Y[p] : 0.0f;
        minx[u] = 0.0f; maxx[u] = 1.0f;
        miny[u] = 0.0f; maxy[u] = 1.0f;
        idx[u] = 0;
    }

    for (int d = 0; d < md; ++d) {
        #pragma unroll
        for (int u = 0; u < PPT; ++u) {
            uint2 nd = nodes[idx[u]];
            float s     = __uint_as_float(nd.x);
            unsigned mt = nd.y;
            float sx = fmaf(s, maxx[u] - minx[u], minx[u]);
            float sy = fmaf(s, maxy[u] - miny[u], miny[u]);
            bool horiz = (mt & 0x10000u) == 0u;
            bool il    = horiz ? (x[u] < sx) : (y[u] < sy);
            if (horiz) { if (il) maxx[u] = sx; else minx[u] = sx; }
            else       { if (il) maxy[u] = sy; else miny[u] = sy; }
            idx[u] = (int)(il ? ((mt >> 8) & 0xFFu) : (mt & 0xFFu));
        }
    }

    #pragma unroll
    for (int u = 0; u < PPT; ++u) {
        int p = base + u * TPB;
        if (p < n) out[p] = svals[idx[u]];
    }
}

extern "C" void kernel_launch(void* const* d_in, const int* in_sizes, int n_in,
                              void* d_out, int out_size, void* d_ws, size_t ws_size,
                              hipStream_t stream) {
    const float* X    = (const float*)d_in[0];  // x_positions [NPX*NPY]
    const float* Y    = (const float*)d_in[1];  // y_positions
    const float* sp   = (const float*)d_in[2];  // split_points [NVD]
    const float* sd   = (const float*)d_in[3];  // split_directions [NVD]
    const float* vals = (const float*)d_in[4];  // values [NVD]
    const float* Lm   = (const float*)d_in[5];  // left_matrix [NVD*NVD]
    const float* Rm   = (const float*)d_in[6];  // right_matrix
    const int*   md   = (const int*)d_in[7];    // max_depth (scalar)

    int* children = (int*)d_ws;                 // [2][NVD] ints

    child_idx_wmma<<<1, 32, 0, stream>>>(Lm, Rm, children);

    const int n = in_sizes[0];                  // 2048*2048
    const int blocks = (n + TPB * PPT - 1) / (TPB * PPT);
    fractal_traverse<<<blocks, TPB, 0, stream>>>(X, Y, sp, sd, vals, children, md,
                                                 (float*)d_out, n);
}
